// HybridQuanvolutionAutoencoder_65481071407114
// MI455X (gfx1250) — matrix operand order repaired and checked by reading, not compiled, not run on hardware
//
#include <hip/hip_runtime.h>

// ---------------------------------------------------------------------------
// Types / WMMA helpers (CDNA5 gfx1250, wave32)
// ---------------------------------------------------------------------------
typedef __attribute__((ext_vector_type(16))) _Float16 v16h;
typedef __attribute__((ext_vector_type(8)))  float    v8f;

union Frag {
  v16h     v;
  uint32_t u[8];
  _Float16 h[16];
};

__device__ __forceinline__ v8f wmma_f16(v16h a, v16h b, v8f c) {
  // D = A(16x32 f16) * B(32x16 f16) + C(16x16 f32)
  return __builtin_amdgcn_wmma_f32_16x16x32_f16(false, a, false, b, (short)0, c,
                                                false, false);
}

__device__ __forceinline__ v8f vzero8() {
  v8f z;
#pragma unroll
  for (int i = 0; i < 8; ++i) z[i] = 0.f;
  return z;
}

// ds_swizzle_b32, group-of-32 mode: lane' = (lane & 0x1f) ^ XORMASK (single DS op,
// no index VGPR / bounds fixup like ds_bpermute).
template <int XORMASK>
__device__ __forceinline__ uint32_t swz_u32(uint32_t x) {
  return (uint32_t)__builtin_amdgcn_ds_swizzle((int)x, (XORMASK << 10) | 0x1f);
}
template <int XORMASK>
__device__ __forceinline__ float swz_f32(float x) {
  return __int_as_float(
      __builtin_amdgcn_ds_swizzle(__float_as_int(x), (XORMASK << 10) | 0x1f));
}

// A fragment: 16x32 f16 tile from row-major [*, ld] at A (element pointer).
// ISA layout: lanes 0-15 hold K=0..7 (V0-V3) and K=16..23 (V4-V7) of row M=lane;
//             lanes 16-31 hold K=8..15 and K=24..31 of row M=lane-16.
__device__ __forceinline__ v16h load_a(const _Float16* A, int ld, int lane) {
  const int m  = lane & 15;
  const int ks = (lane >> 4) * 8;
  const uint32_t* p0 = (const uint32_t*)(A + m * ld + ks);
  const uint32_t* p1 = (const uint32_t*)(A + m * ld + 16 + ks);
  Frag f;
#pragma unroll
  for (int j = 0; j < 4; ++j) { f.u[j] = p0[j]; f.u[4 + j] = p1[j]; }
  return f.v;
}

// B fragment: B[k][n] = W[n0+n][k0+k] (i.e. B = W^T tile), W row-major [*, ld].
// ISA layout: lanes 0-15 hold K=0..7 and K=16..23 of column N=lane;
//             lanes 16-31 hold K=8..15 and K=24..31.
__device__ __forceinline__ v16h load_b(const _Float16* W, int ld, int n0, int k0,
                                       int lane) {
  const int n  = lane & 15;
  const int kk = k0 + ((lane >> 4) << 3);
  const uint32_t* p0 = (const uint32_t*)(W + (n0 + n) * ld + kk);
  const uint32_t* p1 = (const uint32_t*)(W + (n0 + n) * ld + kk + 16);
  Frag f;
#pragma unroll
  for (int j = 0; j < 4; ++j) { f.u[j] = p0[j]; f.u[4 + j] = p1[j]; }
  return f.v;
}

// C/D layout: element (M,N) at lane = (N | ((M>=8)<<4)), vgpr = M & 7.
__device__ __forceinline__ void store_c_f16(_Float16* D, int ld, int n0, v8f c,
                                            const float* bias, bool relu, int lane) {
  const int n  = lane & 15;
  const int mo = (lane >> 4) * 8;
  const float bv = bias[n0 + n];
#pragma unroll
  for (int v = 0; v < 8; ++v) {
    float xv = c[v] + bv;
    if (relu) xv = fmaxf(xv, 0.f);
    D[(mo + v) * ld + n0 + n] = (_Float16)xv;
  }
}

__device__ __forceinline__ void store_c_f32(float* D, int ld, int n0, v8f c,
                                            const float* bias, int lane) {
  const int n  = lane & 15;
  const int mo = (lane >> 4) * 8;
  const float bv = bias[n0 + n];
#pragma unroll
  for (int v = 0; v < 8; ++v) D[(mo + v) * ld + n0 + n] = c[v] + bv;
}

#define LDS_FENCE() asm volatile("s_wait_dscnt 0x0" ::: "memory")

// ---------------------------------------------------------------------------
// Problem constants
// ---------------------------------------------------------------------------
#define BATCH 8192
#define NPATCH 196
#define KPAD 800      // 784 padded to multiple of 32

// ---------------------------------------------------------------------------
// Kernel 0: weight f32->f16 conversion (+ W1 K-pad, + feature pad-column zero)
// ---------------------------------------------------------------------------
__global__ void prep_kernel(const float* __restrict__ W1, const float* __restrict__ W2,
                            const float* __restrict__ W3, const float* __restrict__ W4,
                            const float* __restrict__ W5, const float* __restrict__ W6,
                            _Float16* __restrict__ W1p, _Float16* __restrict__ W2f,
                            _Float16* __restrict__ W3f, _Float16* __restrict__ W4f,
                            _Float16* __restrict__ W5f, _Float16* __restrict__ W6f,
                            _Float16* __restrict__ F) {
  int i = blockIdx.x * blockDim.x + threadIdx.x;
  if (i < 128 * KPAD) {                       // W1 padded [128][800]
    int r = i / KPAD, c = i - r * KPAD;
    W1p[i] = (_Float16)(c < 784 ? W1[r * 784 + c] : 0.f);
    return;
  }
  i -= 128 * KPAD;
  if (i < 64 * 128) { W2f[i] = (_Float16)W2[i]; return; }
  i -= 64 * 128;
  if (i < 32 * 64)  { W3f[i] = (_Float16)W3[i]; return; }
  i -= 32 * 64;
  if (i < 64 * 32)  { W4f[i] = (_Float16)W4[i]; return; }
  i -= 64 * 32;
  if (i < 128 * 64) { W5f[i] = (_Float16)W5[i]; return; }
  i -= 128 * 64;
  if (i < 784 * 128){ W6f[i] = (_Float16)W6[i]; return; }
  i -= 784 * 128;
  if (i < BATCH * 16) {                       // zero feature pad cols 784..799
    int r = i >> 4, c = 784 + (i & 15);
    F[r * KPAD + c] = (_Float16)0.f;
  }
}

// ---------------------------------------------------------------------------
// Kernel 1: quanvolution -> features F [8192][800] f16 (cols 0..783 written)
// One wave handles 32 patches = two 16x16x32 WMMA tiles (x re/im).
// ---------------------------------------------------------------------------
__global__ void quanv_kernel(const float* __restrict__ x,
                             const float* __restrict__ U_re,
                             const float* __restrict__ U_im,
                             _Float16* __restrict__ F) {
  const int lane = threadIdx.x & 31;
  const int wave = (blockIdx.x * blockDim.x + threadIdx.x) >> 5;
  const int gb   = wave * 32;                 // first global patch index

  // B fragments for U_re/U_im: B[k][n] = U[n][k], K rows 16..31 are zero.
  Frag bre, bim;
#pragma unroll
  for (int j = 0; j < 8; ++j) { bre.u[j] = 0; bim.u[j] = 0; }
  {
    const int bn = lane & 15, bk = (lane >> 4) * 8;
#pragma unroll
    for (int k = 0; k < 8; ++k) {
      bre.h[k] = (_Float16)U_re[bn * 16 + bk + k];
      bim.h[k] = (_Float16)U_im[bn * 16 + bk + k];
    }
  }

  // Per-lane patch -> 16-element product state.
  const int g  = gb + lane;
  const int b  = g / NPATCH;
  const int p  = g - b * NPATCH;
  const int pr = p / 14;
  const int pc = p - pr * 14;
  const float* img = x + b * 784 + (2 * pr) * 28 + 2 * pc;
  const float t0 = img[0], t1 = img[1], t2 = img[28], t3 = img[29];
  float c0, s0, c1, s1, c2, s2, c3, s3;
  __sincosf(t0 * 0.5f, &s0, &c0);
  __sincosf(t1 * 0.5f, &s1, &c1);
  __sincosf(t2 * 0.5f, &s2, &c2);
  __sincosf(t3 * 0.5f, &s3, &c3);
  const float e01[4] = { c0 * c1, c0 * s1, s0 * c1, s0 * s1 };
  const float e23[4] = { c2 * c3, c2 * s3, s2 * c3, s2 * s3 };

  uint32_t pk[8];                             // state packed as 8 dwords (f16x2)
#pragma unroll
  for (int j = 0; j < 8; ++j) {
    const int nlo = 2 * j, nhi = 2 * j + 1;
    union { _Float16 h[2]; uint32_t u; } t;
    t.h[0] = (_Float16)(e01[nlo >> 2] * e23[nlo & 3]);
    t.h[1] = (_Float16)(e01[nhi >> 2] * e23[nhi & 3]);
    pk[j] = t.u;
  }
  // Cross-half exchange (lane ^ 16) so both 16-patch A tiles get split-K layout.
  uint32_t opk[8];
#pragma unroll
  for (int j = 0; j < 8; ++j) opk[j] = swz_u32<16>(pk[j]);

  const bool hi = (lane >= 16);
  Frag a0f, a1f;
#pragma unroll
  for (int j = 0; j < 4; ++j) {
    a0f.u[j]     = hi ? opk[j + 4] : pk[j];   // patches gb+0..15
    a1f.u[j]     = hi ? pk[j + 4]  : opk[j];  // patches gb+16..31
    a0f.u[j + 4] = 0;                         // K=16..31 zero pad
    a1f.u[j + 4] = 0;
  }

  const v8f z = vzero8();
  v8f cre0 = wmma_f16(a0f.v, bre.v, z);
  v8f cim0 = wmma_f16(a0f.v, bim.v, z);
  v8f cre1 = wmma_f16(a1f.v, bre.v, z);
  v8f cim1 = wmma_f16(a1f.v, bim.v, z);

  // probs = |psi|^2; the 4 Z-measurements are Walsh coefficients 8,4,2,1:
  // 4-stage signed butterfly, one ds_swizzle + one fma per stage.
  const int nn = lane & 15;
  const int halfsel = lane >> 4;
  const int q = (nn == 8) ? 0 : (nn == 4) ? 1 : (nn == 2) ? 2 : (nn == 1) ? 3 : -1;
  const float sg0 = (nn & 1) ? -1.f : 1.f;
  const float sg1 = (nn & 2) ? -1.f : 1.f;
  const float sg2 = (nn & 4) ? -1.f : 1.f;
  const float sg3 = (nn & 8) ? -1.f : 1.f;
#pragma unroll
  for (int t = 0; t < 2; ++t) {
    const v8f cre = t ? cre1 : cre0;
    const v8f cim = t ? cim1 : cim0;
#pragma unroll
    for (int v = 0; v < 8; ++v) {
      float pv = cre[v] * cre[v] + cim[v] * cim[v];
      pv = fmaf(sg0, pv, swz_f32<1>(pv));
      pv = fmaf(sg1, pv, swz_f32<2>(pv));
      pv = fmaf(sg2, pv, swz_f32<4>(pv));
      pv = fmaf(sg3, pv, swz_f32<8>(pv));
      if (q >= 0) {
        const int gg  = gb + t * 16 + v + 8 * halfsel;   // patch index
        const int bb  = gg / NPATCH;
        const int col = (gg - bb * NPATCH) * 4 + q;      // p*4+q in 0..783
        F[bb * KPAD + col] = (_Float16)pv;
      }
    }
  }
}

// ---------------------------------------------------------------------------
// Kernel 2: H1 = relu(F @ W1^T + b1)   [8192,800]x[128,800] -> [8192,128] f16
// Block = 8 waves x 16 rows; W1 K-slab (128x32, 8KB) staged in LDS per K step
// so B is read from L2 once per block instead of once per wave.
// ---------------------------------------------------------------------------
__global__ void gemm1_kernel(const _Float16* __restrict__ F,
                             const _Float16* __restrict__ W1p,
                             const float* __restrict__ b1,
                             _Float16* __restrict__ H1) {
  __shared__ _Float16 sB[128 * 32];           // 8 KB
  const int lane  = threadIdx.x & 31;
  const int wslot = threadIdx.x >> 5;
  const int row0  = (blockIdx.x * 8 + wslot) * 16;

  v8f acc[8];
#pragma unroll
  for (int i = 0; i < 8; ++i) acc[i] = vzero8();

  for (int k = 0; k < KPAD; k += 32) {
    __syncthreads();                          // previous slab fully consumed
    uint32_t* dst = (uint32_t*)sB;
    for (int d = threadIdx.x; d < 2048; d += 256) {
      const int r = d >> 4, c = d & 15;       // [128 rows][16 dwords]
      dst[d] = ((const uint32_t*)(W1p + r * KPAD + k))[c];
    }
    __syncthreads();
    const v16h a = load_a(F + row0 * KPAD + k, KPAD, lane);
#pragma unroll
    for (int nt = 0; nt < 8; ++nt)
      acc[nt] = wmma_f16(a, load_b(sB, 32, nt * 16, 0, lane), acc[nt]);
  }
#pragma unroll
  for (int nt = 0; nt < 8; ++nt)
    store_c_f16(H1 + row0 * 128, 128, nt * 16, acc[nt], b1, true, lane);
}

// ---------------------------------------------------------------------------
// Kernel 3: fused layers 2..5 (128->64->32->64->128), per-wave LDS ping-pong.
// ---------------------------------------------------------------------------
__global__ void mlp_mid_kernel(const _Float16* __restrict__ H1,
                               const _Float16* __restrict__ W2f,
                               const _Float16* __restrict__ W3f,
                               const _Float16* __restrict__ W4f,
                               const _Float16* __restrict__ W5f,
                               const float* __restrict__ b2,
                               const float* __restrict__ b3,
                               const float* __restrict__ b4,
                               const float* __restrict__ b5,
                               _Float16* __restrict__ H4) {
  __shared__ _Float16 sbuf[8][2][16 * 64];    // 32 KB: per-wave ping/pong tiles
  const int lane  = threadIdx.x & 31;
  const int wslot = threadIdx.x >> 5;
  const int wave  = (blockIdx.x * blockDim.x + threadIdx.x) >> 5;
  const int row0  = wave * 16;
  _Float16* bufA = sbuf[wslot][0];
  _Float16* bufB = sbuf[wslot][1];

  // Layer 2: h2 = relu(h1 @ W2^T + b2)  K=128 N=64  (global -> bufA)
  {
    v16h a[4];
#pragma unroll
    for (int kc = 0; kc < 4; ++kc)
      a[kc] = load_a(H1 + row0 * 128 + kc * 32, 128, lane);
#pragma unroll
    for (int nt = 0; nt < 4; ++nt) {
      v8f acc = vzero8();
#pragma unroll
      for (int kc = 0; kc < 4; ++kc)
        acc = wmma_f16(a[kc], load_b(W2f, 128, nt * 16, kc * 32, lane), acc);
      store_c_f16(bufA, 64, nt * 16, acc, b2, true, lane);
    }
  }
  LDS_FENCE();
  // Layer 3: z = h2 @ W3^T + b3  K=64 N=32 (no relu)  (bufA -> bufB)
  {
    const v16h a0 = load_a(bufA, 64, lane);
    const v16h a1 = load_a(bufA + 32, 64, lane);
#pragma unroll
    for (int nt = 0; nt < 2; ++nt) {
      v8f acc = vzero8();
      acc = wmma_f16(a0, load_b(W3f, 64, nt * 16, 0, lane), acc);
      acc = wmma_f16(a1, load_b(W3f, 64, nt * 16, 32, lane), acc);
      store_c_f16(bufB, 32, nt * 16, acc, b3, false, lane);
    }
  }
  LDS_FENCE();
  // Layer 4: h3 = relu(z @ W4^T + b4)  K=32 N=64  (bufB -> bufA)
  {
    const v16h a = load_a(bufB, 32, lane);
#pragma unroll
    for (int nt = 0; nt < 4; ++nt) {
      v8f acc = wmma_f16(a, load_b(W4f, 32, nt * 16, 0, lane), vzero8());
      store_c_f16(bufA, 64, nt * 16, acc, b4, true, lane);
    }
  }
  LDS_FENCE();
  // Layer 5: h4 = relu(h3 @ W5^T + b5)  K=64 N=128  (bufA -> global H4)
  {
    const v16h a0 = load_a(bufA, 64, lane);
    const v16h a1 = load_a(bufA + 32, 64, lane);
#pragma unroll
    for (int nt = 0; nt < 8; ++nt) {
      v8f acc = vzero8();
      acc = wmma_f16(a0, load_b(W5f, 64, nt * 16, 0, lane), acc);
      acc = wmma_f16(a1, load_b(W5f, 64, nt * 16, 32, lane), acc);
      store_c_f16(H4 + row0 * 128, 128, nt * 16, acc, b5, true, lane);
    }
  }
}

// ---------------------------------------------------------------------------
// Kernel 4: out = H4 @ W6^T + b6   [8192,128]x[784,128] -> [8192,784] f32
// Block = 8 waves x 16 rows; W6 N-tile (16x128, 4KB) staged in LDS per nt.
// ---------------------------------------------------------------------------
__global__ void gemm6_kernel(const _Float16* __restrict__ H4,
                             const _Float16* __restrict__ W6f,
                             const float* __restrict__ b6,
                             float* __restrict__ Out) {
  __shared__ _Float16 sB[16 * 128];           // 4 KB
  const int lane  = threadIdx.x & 31;
  const int wslot = threadIdx.x >> 5;
  const int row0  = (blockIdx.x * 8 + wslot) * 16;

  v16h a[4];
#pragma unroll
  for (int kc = 0; kc < 4; ++kc)
    a[kc] = load_a(H4 + row0 * 128 + kc * 32, 128, lane);

  for (int nt = 0; nt < 49; ++nt) {
    __syncthreads();                          // previous tile fully consumed
    uint32_t* dst = (uint32_t*)sB;
    const uint32_t* src = (const uint32_t*)(W6f + nt * 16 * 128);
    for (int d = threadIdx.x; d < 1024; d += 256) dst[d] = src[d];
    __syncthreads();
    v8f acc = vzero8();
#pragma unroll
    for (int kc = 0; kc < 4; ++kc)
      acc = wmma_f16(a[kc], load_b(sB, 128, 0, kc * 32, lane), acc);
    store_c_f32(Out + row0 * 784, 784, nt * 16, acc, b6, lane);
  }
}

// ---------------------------------------------------------------------------
// Host launch
// ---------------------------------------------------------------------------
extern "C" void kernel_launch(void* const* d_in, const int* in_sizes, int n_in,
                              void* d_out, int out_size, void* d_ws, size_t ws_size,
                              hipStream_t stream) {
  const float* x    = (const float*)d_in[0];
  const float* U_re = (const float*)d_in[1];
  const float* U_im = (const float*)d_in[2];
  const float* W1 = (const float*)d_in[3];  const float* b1 = (const float*)d_in[4];
  const float* W2 = (const float*)d_in[5];  const float* b2 = (const float*)d_in[6];
  const float* W3 = (const float*)d_in[7];  const float* b3 = (const float*)d_in[8];
  const float* W4 = (const float*)d_in[9];  const float* b4 = (const float*)d_in[10];
  const float* W5 = (const float*)d_in[11]; const float* b5 = (const float*)d_in[12];
  const float* W6 = (const float*)d_in[13]; const float* b6 = (const float*)d_in[14];
  float* out = (float*)d_out;

  _Float16* ws  = (_Float16*)d_ws;
  _Float16* F   = ws;                          // 8192*800
  _Float16* W1p = F   + (size_t)BATCH * KPAD;  // 128*800
  _Float16* W2f = W1p + 128 * KPAD;            // 64*128
  _Float16* W3f = W2f + 64 * 128;              // 32*64
  _Float16* W4f = W3f + 32 * 64;               // 64*32
  _Float16* W5f = W4f + 64 * 32;               // 128*64
  _Float16* W6f = W5f + 128 * 64;              // 784*128
  _Float16* H1  = W6f + 784 * 128;             // 8192*128
  _Float16* H4  = H1  + (size_t)BATCH * 128;   // 8192*128

  const int prep_elems = 128 * KPAD + 64 * 128 + 32 * 64 + 64 * 32 + 128 * 64 +
                         784 * 128 + BATCH * 16;
  prep_kernel<<<(prep_elems + 255) / 256, 256, 0, stream>>>(
      W1, W2, W3, W4, W5, W6, W1p, W2f, W3f, W4f, W5f, W6f, F);

  // 8192*196 patches / 32 per wave = 50176 waves; 8 waves per 256-thread block.
  quanv_kernel<<<50176 / 8, 256, 0, stream>>>(x, U_re, U_im, F);

  // 8192/16 = 512 waves -> 64 blocks of 8 waves for each GEMM stage.
  gemm1_kernel<<<64, 256, 0, stream>>>(F, W1p, b1, H1);
  mlp_mid_kernel<<<64, 256, 0, stream>>>(H1, W2f, W3f, W4f, W5f, b2, b3, b4, b5, H4);
  gemm6_kernel<<<64, 256, 0, stream>>>(H4, W6f, b6, out);
}
